// HierarchicalModularLoss_86912958202423
// MI455X (gfx1250) — matrix operand order repaired and checked by reading, not compiled, not run on hardware
//
#include <hip/hip_runtime.h>
#include <hip/hip_bf16.h>

typedef __attribute__((ext_vector_type(16))) __bf16 v16bf;
typedef __attribute__((ext_vector_type(8)))  float  v8f;

#define NROWS   8192
#define DDIM    256
#define NUMIDS  2048
#define BLK64   (NROWS / 64)   // 128 64-row blocks per matrix edge

// d_ws layout:
//   [0,128)                     float acc[32]:
//       acc[0]=primary_sum  acc[1]=aux_sum
//       acc[2+2p]=pos_sum(pair p)  acc[3+2p]=neg_sum(pair p)  acc[8]=sum(c_k^2)
//   [128, 128+8192)             uint hist[2048]
//   [8320, 8320+3*8192*256*2)   bf16 normalized embeddings (row-major, 3 matrices)

__global__ void k_zero(float* acc, unsigned* hist) {
    int t = threadIdx.x;
    if (t < 32) acc[t] = 0.0f;
    for (int i = t; i < NUMIDS; i += blockDim.x) hist[i] = 0u;
}

__global__ void k_scalar(const float* __restrict__ outs, const float* __restrict__ tgts,
                         const float* __restrict__ m0, const float* __restrict__ m1,
                         const float* __restrict__ m2, const int* __restrict__ ids,
                         float* acc, unsigned* hist) {
    __shared__ float sp[256];
    __shared__ float sa[256];
    int i = blockIdx.x * blockDim.x + threadIdx.x;
    float p = 0.0f, a = 0.0f;
    if (i < NROWS) {
        float o = outs[i];
        float d = o - tgts[i];
        p = d * d;
        float d0 = m0[i] - o, d1 = m1[i] - o, d2 = m2[i] - o;
        a = d0 * d0 + d1 * d1 + d2 * d2;
        atomicAdd(&hist[ids[i]], 1u);
    }
    sp[threadIdx.x] = p; sa[threadIdx.x] = a;
    __syncthreads();
    for (int s = 128; s > 0; s >>= 1) {
        if (threadIdx.x < s) {
            sp[threadIdx.x] += sp[threadIdx.x + s];
            sa[threadIdx.x] += sa[threadIdx.x + s];
        }
        __syncthreads();
    }
    if (threadIdx.x == 0) { atomicAdd(&acc[0], sp[0]); atomicAdd(&acc[1], sa[0]); }
}

__global__ void k_npos(const unsigned* __restrict__ hist, float* acc) {
    __shared__ float s[256];
    float v = 0.0f;
    for (int i = threadIdx.x; i < NUMIDS; i += blockDim.x) {
        float c = (float)hist[i];
        v += c * c;
    }
    s[threadIdx.x] = v;
    __syncthreads();
    for (int st = 128; st > 0; st >>= 1) {
        if (threadIdx.x < st) s[threadIdx.x] += s[threadIdx.x + st];
        __syncthreads();
    }
    if (threadIdx.x == 0) acc[8] = s[0];
}

// One block (256 threads == DDIM) per row; l2-normalize and RNE-convert to bf16.
__global__ void k_norm(const float* __restrict__ e0, const float* __restrict__ e1,
                       const float* __restrict__ e2, unsigned short* __restrict__ zbuf) {
    __shared__ float s[256];
    int row = blockIdx.x;                 // 0 .. 3*NROWS-1
    int mod = row / NROWS;
    int r   = row - mod * NROWS;
    const float* src = (mod == 0) ? e0 : (mod == 1) ? e1 : e2;
    float x = src[(size_t)r * DDIM + threadIdx.x];
    s[threadIdx.x] = x * x;
    __syncthreads();
    for (int st = 128; st > 0; st >>= 1) {
        if (threadIdx.x < st) s[threadIdx.x] += s[threadIdx.x + st];
        __syncthreads();
    }
    float z = x * rsqrtf(s[0]);
    unsigned u = __float_as_uint(z);
    unsigned h = (u + 0x7FFFu + ((u >> 16) & 1u)) >> 16;   // round-to-nearest-even bf16
    zbuf[(size_t)row * DDIM + threadIdx.x] = (unsigned short)h;
}

// Masked GEMM-reduce: sim = (Z_i Z_j^T)/TEMP. Each wave computes a 64x64 output
// block as a 4x4 grid of 16x16 WMMA tiles: per K=32 step, 8 fragment loads feed
// 16 WMMAs (1.0 b128 load per WMMA). Epilogue applies pos/neg masks branchlessly.
__global__ __launch_bounds__(256) void k_gemm(const unsigned short* __restrict__ zbuf,
                                              const int* __restrict__ ids,
                                              float* acc) {
    const int pair = blockIdx.y;                 // 0:(0,1) 1:(0,2) 2:(1,2)
    const int pi = (pair == 2) ? 1 : 0;
    const int pj = (pair == 0) ? 1 : 2;
    const unsigned short* A = zbuf + (size_t)pi * NROWS * DDIM;
    const unsigned short* B = zbuf + (size_t)pj * NROWS * DDIM;

    const int lane    = threadIdx.x & 31;
    const int halfsel = lane >> 4;               // 0 for lanes 0-15, 1 for 16-31
    const int l16     = lane & 15;
    const int gwave   = blockIdx.x * (blockDim.x >> 5) + (threadIdx.x >> 5);
    const int nwaves  = gridDim.x * (blockDim.x >> 5);

    float pos_sum = 0.0f, neg_sum = 0.0f;

    for (int t = gwave; t < BLK64 * BLK64; t += nwaves) {
        const int bm = t >> 7;                   // / 128
        const int bn = t & 127;                  // % 128

        const unsigned short* arow[4];
        const unsigned short* brow[4];
        #pragma unroll
        for (int i = 0; i < 4; ++i) {
            arow[i] = A + (size_t)(bm * 64 + 16 * i + l16) * DDIM;
            brow[i] = B + (size_t)(bn * 64 + 16 * i + l16) * DDIM;
        }

        v8f cc[4][4];
        #pragma unroll
        for (int i = 0; i < 4; ++i)
            #pragma unroll
            for (int j = 0; j < 4; ++j)
                cc[i][j] = (v8f){0.f,0.f,0.f,0.f,0.f,0.f,0.f,0.f};

        #pragma unroll
        for (int kb = 0; kb < DDIM; kb += 32) {
            union F { v16bf v; uint4 q[2]; };
            F fa[4], fb[4];
            // A 16x32 fragment: lanes 0-15 hold K kb..kb+7 then kb+16..kb+23;
            //                   lanes 16-31 hold K kb+8..kb+15 then kb+24..kb+31.
            const int ka0 = kb + halfsel * 8;
            // B 32x16 fragment: same K order as A (lanes 0-15 kb..kb+7, kb+16..kb+23; lanes 16-31 kb+8..kb+15, kb+24..kb+31).
            const int kb0 = kb + halfsel * 8;
            #pragma unroll
            for (int i = 0; i < 4; ++i) {
                fa[i].q[0] = *(const uint4*)(arow[i] + ka0);
                fa[i].q[1] = *(const uint4*)(arow[i] + ka0 + 16);
                const uint4* bp = (const uint4*)(brow[i] + kb0);
                fb[i].q[0] = bp[0];
                fb[i].q[1] = bp[2];
            }
            #pragma unroll
            for (int i = 0; i < 4; ++i)
                #pragma unroll
                for (int j = 0; j < 4; ++j)
                    cc[i][j] = __builtin_amdgcn_wmma_f32_16x16x32_bf16(
                        false, fa[i].v, false, fb[j].v, (short)0, cc[i][j], false, false);
        }

        // ids for the four 8-row m-strips this lane owns (contiguous -> int4 loads)
        int idm[4][8];
        int mb[4];
        #pragma unroll
        for (int i = 0; i < 4; ++i) {
            mb[i] = bm * 64 + 16 * i + 8 * halfsel;
            int4 lo = *(const int4*)(ids + mb[i]);
            int4 hi = *(const int4*)(ids + mb[i] + 4);
            idm[i][0] = lo.x; idm[i][1] = lo.y; idm[i][2] = lo.z; idm[i][3] = lo.w;
            idm[i][4] = hi.x; idm[i][5] = hi.y; idm[i][6] = hi.z; idm[i][7] = hi.w;
        }
        int nn[4], idn[4];
        #pragma unroll
        for (int j = 0; j < 4; ++j) {
            nn[j]  = bn * 64 + 16 * j + l16;
            idn[j] = ids[nn[j]];
        }

        #pragma unroll
        for (int i = 0; i < 4; ++i)
            #pragma unroll
            for (int j = 0; j < 4; ++j)
                #pragma unroll
                for (int v = 0; v < 8; ++v) {
                    // branchless: ternary selects -> v_cndmask, no exec divergence
                    float s = cc[i][j][v] * 2.0f;                 // sim / TEMP
                    bool pos = (idm[i][v] == idn[j]) & ((mb[i] + v) != nn[j]);
                    pos_sum += pos ? s : 0.0f;
                    neg_sum += pos ? 0.0f : fmaxf(1.0f + s, 0.0f);   // MARGIN=1
                }
    }

    #pragma unroll
    for (int off = 16; off > 0; off >>= 1) {
        pos_sum += __shfl_xor(pos_sum, off, 32);
        neg_sum += __shfl_xor(neg_sum, off, 32);
    }
    if (lane == 0) {
        atomicAdd(&acc[2 + 2 * pair], pos_sum);
        atomicAdd(&acc[3 + 2 * pair], neg_sum);
    }
}

__global__ void k_final(const float* __restrict__ acc, float* __restrict__ out) {
    if (threadIdx.x == 0 && blockIdx.x == 0) {
        const float Nf = (float)NROWS;
        float primary = acc[0] / Nf;
        float aux     = acc[1] / (3.0f * Nf);
        float sumsq   = acc[8];
        float n_pos   = sumsq - Nf;                 // same-id off-diagonal pairs
        float n_neg   = Nf * Nf - n_pos;            // complement (diagonal is negative)
        float contra = 0.0f;
        #pragma unroll
        for (int p = 0; p < 3; ++p)
            contra += (-acc[2 + 2 * p] / n_pos + acc[3 + 2 * p] / n_neg);
        contra *= (1.0f / 3.0f);
        out[0] = primary + 0.1f * aux + 0.05f * contra;
    }
}

extern "C" void kernel_launch(void* const* d_in, const int* in_sizes, int n_in,
                              void* d_out, int out_size, void* d_ws, size_t ws_size,
                              hipStream_t stream) {
    const float* outs = (const float*)d_in[0];
    const float* tgts = (const float*)d_in[1];
    const float* m0   = (const float*)d_in[2];
    const float* m1   = (const float*)d_in[3];
    const float* m2   = (const float*)d_in[4];
    const float* e0   = (const float*)d_in[5];
    const float* e1   = (const float*)d_in[6];
    const float* e2   = (const float*)d_in[7];
    const int*   ids  = (const int*)d_in[8];

    char* ws = (char*)d_ws;
    float*          acc  = (float*)ws;
    unsigned*       hist = (unsigned*)(ws + 128);
    unsigned short* zbuf = (unsigned short*)(ws + 128 + NUMIDS * sizeof(unsigned));

    k_zero  <<<1, 256, 0, stream>>>(acc, hist);
    k_scalar<<<NROWS / 256, 256, 0, stream>>>(outs, tgts, m0, m1, m2, ids, acc, hist);
    k_npos  <<<1, 256, 0, stream>>>(hist, acc);
    k_norm  <<<3 * NROWS, 256, 0, stream>>>(e0, e1, e2, zbuf);
    dim3 g(1024, 3, 1);
    k_gemm  <<<g, 256, 0, stream>>>(zbuf, ids, acc);
    k_final <<<1, 64, 0, stream>>>(acc, (float*)d_out);
}
